// Network_79396765434056
// MI455X (gfx1250) — matrix operand order repaired and checked by reading, not compiled, not run on hardware
//
#include <hip/hip_runtime.h>
#include <hip/hip_bf16.h>
#include <cstdint>

typedef __attribute__((ext_vector_type(16))) __bf16 v16bf;
typedef __attribute__((ext_vector_type(8)))  float  v8f;
typedef unsigned int u32x4 __attribute__((ext_vector_type(4)));

union FragBF { v16bf v; u32x4 q[2]; };

#if defined(__has_builtin)
#  if __has_builtin(__builtin_amdgcn_global_load_async_to_lds_b128)
#    define HAVE_ASYNC_LDS 1
#  endif
#endif
#ifndef HAVE_ASYNC_LDS
#  define HAVE_ASYNC_LDS 0
#endif

// Builtin signature (from hipcc diagnostic): param0 is
//   __attribute__((__vector_size__(16))) int __device__ *   (AS1 / global)
// so: (v4i as1* gsrc, v4i as3* ldsdst, imm offset, imm cpol)
typedef int v4i __attribute__((__vector_size__(16)));
typedef __attribute__((address_space(1))) v4i* as1_v4i_p;
typedef __attribute__((address_space(3))) v4i* as3_v4i_p;

// ---------- helpers ----------
static __device__ __forceinline__ unsigned short f2bf(float f) {
    unsigned u = __float_as_uint(f);
    unsigned r = (u + 0x7FFFu + ((u >> 16) & 1u)) >> 16;   // round-to-nearest-even
    return (unsigned short)r;
}
static __device__ __forceinline__ float sigm(float x) {
    return 1.0f / (1.0f + __expf(-x));
}

// 16-byte global -> LDS copy (async on CDNA5 if toolchain exposes the builtin)
static __device__ __forceinline__ void copy16_g2l(const unsigned short* g, unsigned short* l) {
#if HAVE_ASYNC_LDS
    // LDS aperture keeps the byte offset in addr[31:0]; AS1 value == flat value.
    __builtin_amdgcn_global_load_async_to_lds_b128(
        (as1_v4i_p)(uintptr_t)g, (as3_v4i_p)(unsigned)(uintptr_t)l, 0, 0);
#else
    *(u32x4*)l = *(const u32x4*)g;
#endif
}
static __device__ __forceinline__ void wait_async_lds() {
#if defined(__has_builtin) && __has_builtin(__builtin_amdgcn_s_wait_asynccnt)
    __builtin_amdgcn_s_wait_asynccnt(0);
#else
    asm volatile("s_wait_asynccnt 0x0" ::: "memory");
#endif
}

// ---------- kernel: zero scalar output ----------
__global__ void zero_out_kernel(float* out) { out[0] = 0.0f; }

// ---------- kernel: embedding gather -> bf16 x (512 x 256, cols 250..255 zero) ----------
__global__ void embed_kernel(const int* __restrict__ wi, const int* __restrict__ ti,
                             const float* __restrict__ Ww, const float* __restrict__ Wt,
                             unsigned short* __restrict__ xbf) {
    int idx = blockIdx.x * blockDim.x + threadIdx.x;
    if (idx >= 512 * 256) return;
    int t = idx >> 8, c = idx & 255;
    float v = 0.0f;
    if (c < 200)       v = Ww[(size_t)wi[t] * 200 + c];
    else if (c < 250)  v = Wt[(size_t)ti[t] * 50 + (c - 200)];
    xbf[idx] = f2bf(v);
}

// ---------- kernel: f32 (rows x ks) -> bf16 (rows x kd), zero-pad cols >= ks ----------
__global__ void convpad_kernel(const float* __restrict__ src, unsigned short* __restrict__ dst,
                               int rows, int ks, int kd) {
    int idx = blockIdx.x * blockDim.x + threadIdx.x;
    if (idx >= rows * kd) return;
    int r = idx / kd, c = idx - r * kd;
    float v = (c < ks) ? src[(size_t)r * ks + c] : 0.0f;
    dst[idx] = f2bf(v);
}

// ---------- kernel: transpose f32 (R x C) -> (C x R) ----------
__global__ void transpose_kernel(const float* __restrict__ src, float* __restrict__ dst,
                                 int R, int C) {
    int idx = blockIdx.x * blockDim.x + threadIdx.x;
    if (idx >= R * C) return;
    int r = idx / C, c = idx - r * C;
    dst[(size_t)c * R + r] = src[idx];
}

// ---------- kernel: bf16 WMMA GEMM  C[M,N] = A[M,K] * Bw[N,K]^T + bias (+relu) ----------
// block = 256 threads = 8 waves; wave tile 32x64; block tile 128(M) x 128(N).
// A/B tiles double-buffered in LDS, filled with async global->LDS copies.
__global__ __launch_bounds__(256)
void gemm_bf16_kernel(const unsigned short* __restrict__ A,
                      const unsigned short* __restrict__ Bw,
                      const float* __restrict__ bias,
                      float* __restrict__ C,
                      int M, int N, int K, int relu) {
    __shared__ __align__(16) unsigned short sA[2][128 * 32];
    __shared__ __align__(16) unsigned short sB[2][128 * 32];

    const int tid  = threadIdx.x;
    const int lane = tid & 31;
    const int wave = tid >> 5;
    const int wm   = wave >> 1;          // 0..3
    const int wn   = wave & 1;           // 0..1
    const int half = lane >> 4;          // 0/1
    const int l16  = lane & 15;

    const int mBlk = blockIdx.y * 128;
    const int nBlk = blockIdx.x * 128;

    v8f acc[2][4] = {};
    const int nk = K >> 5;

    auto stage = [&](int kt, int b) {
        const int k0 = kt << 5;
#pragma unroll
        for (int c = tid; c < 512; c += 256) {       // 512 chunks of 8 ushorts per tile
            int row = c >> 2;
            int col = (c & 3) << 3;
            copy16_g2l(A  + (size_t)(mBlk + row) * K + k0 + col, &sA[b][row * 32 + col]);
            copy16_g2l(Bw + (size_t)(nBlk + row) * K + k0 + col, &sB[b][row * 32 + col]);
        }
    };

    stage(0, 0);
    int buf = 0;
    for (int kt = 0; kt < nk; ++kt) {
        wait_async_lds();
        __syncthreads();                  // tile kt resident in sA[buf]/sB[buf]
        if (kt + 1 < nk) stage(kt + 1, buf ^ 1);   // prefetch next tile while computing

        FragBF a[2];
#pragma unroll
        for (int i = 0; i < 2; ++i) {
            const unsigned short* pa = &sA[buf][(wm * 32 + i * 16 + l16) * 32 + half * 8];
            a[i].q[0] = *(const u32x4*)(pa);
            a[i].q[1] = *(const u32x4*)(pa + 16);
        }
#pragma unroll
        for (int j = 0; j < 4; ++j) {
            FragBF b;
            const unsigned short* pb = &sB[buf][(wn * 64 + j * 16 + l16) * 32 + half * 8];
            b.q[0] = *(const u32x4*)(pb);
            b.q[1] = *(const u32x4*)(pb + 16);
            acc[0][j] = __builtin_amdgcn_wmma_f32_16x16x32_bf16(
                false, a[0].v, false, b.v, (short)0, acc[0][j], false, false);
            acc[1][j] = __builtin_amdgcn_wmma_f32_16x16x32_bf16(
                false, a[1].v, false, b.v, (short)0, acc[1][j], false, false);
        }
        buf ^= 1;
    }

    const int mBase = mBlk + wm * 32;
    const int nBase = nBlk + wn * 64;
#pragma unroll
    for (int i = 0; i < 2; ++i) {
#pragma unroll
        for (int j = 0; j < 4; ++j) {
            int col = nBase + j * 16 + l16;
            float bv = bias ? bias[col] : 0.0f;
#pragma unroll
            for (int r = 0; r < 8; ++r) {
                int row = mBase + i * 16 + half * 8 + r;
                float v = acc[i][j][r] + bv;
                if (relu) v = v > 0.0f ? v : 0.0f;
                C[(size_t)row * N + col] = v;
            }
        }
    }
}

// ---------- kernel: LSTM recurrence, U=512, gates=2048; block0=fwd dir, block1=bwd dir ----------
__global__ __launch_bounds__(1024)
void lstm_rec2_kernel(const float* __restrict__ xgF, const float* __restrict__ xgB,
                      const float* __restrict__ WTF, const float* __restrict__ WTB,
                      float* __restrict__ houtF32, unsigned short* __restrict__ houtBF,
                      int ldh, int T) {
    __shared__ float hS[512];
    __shared__ float gS[2048];
    const int t = threadIdx.x;
    const bool back = (blockIdx.x == 1);
    const float* xg = back ? xgB : xgF;
    const float* WT = back ? WTB : WTF;
    const int colOff = back ? 512 : 0;

    if (t < 512) hS[t] = 0.0f;
    float c = 0.0f;
    __syncthreads();

    for (int s = 0; s < T; ++s) {
        int tt = back ? (T - 1 - s) : s;
        const float* xr = xg + (size_t)tt * 2048;
        float a0 = xr[t];
        float a1 = xr[t + 1024];
        for (int k = 0; k < 512; ++k) {
            float hk = hS[k];
            a0 = fmaf(hk, WT[(size_t)k * 2048 + t],        a0);
            a1 = fmaf(hk, WT[(size_t)k * 2048 + t + 1024], a1);
        }
        gS[t]        = a0;
        gS[t + 1024] = a1;
        __syncthreads();
        if (t < 512) {
            float gi = gS[t], gf = gS[t + 512], gg = gS[t + 1024], go = gS[t + 1536];
            float i = sigm(gi);
            float f = sigm(gf);
            float g = tanhf(gg);
            float o = sigm(go);
            c = f * c + i * g;
            float h = o * tanhf(c);
            hS[t] = h;
            if (houtF32) houtF32[(size_t)tt * ldh + colOff + t] = h;
            if (houtBF)  houtBF[(size_t)tt * ldh + colOff + t] = f2bf(h);
        }
        __syncthreads();
    }
}

// ---------- kernel: span feature gather -> bf16 H (n x kk*1024) ----------
// outcat: 512 x 1024 (cols 0..511 = fwd, 512..1023 = back)
__global__ void span_gather_kernel(const float* __restrict__ outcat,
                                   const int* __restrict__ lefts,
                                   const int* __restrict__ rights,
                                   unsigned short* __restrict__ H,
                                   int n, int kk) {
    const int width = kk * 1024;
    int idx = blockIdx.x * blockDim.x + threadIdx.x;
    if (idx >= n * width) return;
    int s = idx / width, c = idx - s * width;
    const int halfw = kk * 512;
    float v;
    if (c < halfw) {
        int k = c >> 9, j = c & 511;
        int l = lefts[s * kk + k], r = rights[s * kk + k];
        v = outcat[(size_t)r * 1024 + j] - outcat[(size_t)(l - 1) * 1024 + j];
    } else {
        int c2 = c - halfw;
        int k = c2 >> 9, j = c2 & 511;
        int l = lefts[s * kk + k], r = rights[s * kk + k];
        v = outcat[(size_t)l * 1024 + 512 + j] - outcat[(size_t)(r + 1) * 1024 + 512 + j];
    }
    H[idx] = f2bf(v);
}

// ---------- kernel: output layer + NLL loss, one block per row ----------
__global__ __launch_bounds__(256)
void loss_kernel(const float* __restrict__ Mmid, const float* __restrict__ W2,
                 const float* __restrict__ b2, const int* __restrict__ tgt,
                 float* __restrict__ out, int nout, float scale) {
    __shared__ float row[1024];
    __shared__ float sc[64];
    const int r = blockIdx.x;
    const float* h = Mmid + (size_t)r * 1024;
    for (int k = threadIdx.x; k < 1024; k += 256) row[k] = h[k];
    __syncthreads();
    if ((int)threadIdx.x < nout) {
        float s = b2[threadIdx.x];
        const float* w = W2 + (size_t)threadIdx.x * 1024;
        for (int k = 0; k < 1024; ++k) s = fmaf(row[k], w[k], s);
        sc[threadIdx.x] = s;
    }
    __syncthreads();
    if (threadIdx.x == 0) {
        float m = -1e30f;
        for (int o = 0; o < nout; ++o) m = fmaxf(m, sc[o]);
        float l = 0.0f;
        for (int o = 0; o < nout; ++o) l += __expf(sc[o] - m);
        float lse = m + __logf(l);
        atomicAdd(out, (lse - sc[tgt[r]]) * scale);
    }
}

// =================== host side ===================
extern "C" void kernel_launch(void* const* d_in, const int* in_sizes, int n_in,
                              void* d_out, int out_size, void* d_ws, size_t ws_size,
                              hipStream_t stream) {
    const int*   word_inds   = (const int*)d_in[0];
    const int*   tag_inds    = (const int*)d_in[1];
    const int*   s_targets   = (const int*)d_in[2];
    const int*   l_targets   = (const int*)d_in[3];
    const int*   s_lefts     = (const int*)d_in[4];
    const int*   s_rights    = (const int*)d_in[5];
    const int*   l_lefts     = (const int*)d_in[6];
    const int*   l_rights    = (const int*)d_in[7];
    const float* W_word      = (const float*)d_in[8];
    const float* W_tag       = (const float*)d_in[9];
    const float* Wih_l0f     = (const float*)d_in[10];
    const float* Whh_l0f     = (const float*)d_in[11];
    const float* b_l0f       = (const float*)d_in[12];
    const float* Wih_l0b     = (const float*)d_in[13];
    const float* Whh_l0b     = (const float*)d_in[14];
    const float* b_l0b       = (const float*)d_in[15];
    const float* Wih_l1f     = (const float*)d_in[16];
    const float* Whh_l1f     = (const float*)d_in[17];
    const float* b_l1f       = (const float*)d_in[18];
    const float* Wih_l1b     = (const float*)d_in[19];
    const float* Whh_l1b     = (const float*)d_in[20];
    const float* b_l1b       = (const float*)d_in[21];
    const float* Ws1         = (const float*)d_in[22];
    const float* bs1         = (const float*)d_in[23];
    const float* Ws2         = (const float*)d_in[24];
    const float* bs2         = (const float*)d_in[25];
    const float* Wl1         = (const float*)d_in[26];
    const float* bl1         = (const float*)d_in[27];
    const float* Wl2         = (const float*)d_in[28];
    const float* bl2         = (const float*)d_in[29];
    float* out = (float*)d_out;

    // ---- workspace carve-up ----
    char* ws = (char*)d_ws;
    size_t off = 0;
    auto carve = [&](size_t bytes) -> void* {
        void* p = (void*)(ws + off);
        off += (bytes + 255) & ~(size_t)255;
        return p;
    };
    unsigned short* x_bf   = (unsigned short*)carve((size_t)512 * 256 * 2);
    unsigned short* wslot  = (unsigned short*)carve((size_t)1024 * 4096 * 2); // reused for all converted weights
    float* whhT0f = (float*)carve((size_t)512 * 2048 * 4);
    float* whhT0b = (float*)carve((size_t)512 * 2048 * 4);
    float* whhT1f = (float*)carve((size_t)512 * 2048 * 4);
    float* whhT1b = (float*)carve((size_t)512 * 2048 * 4);
    float* xgA    = (float*)carve((size_t)512 * 2048 * 4);
    float* xgB    = (float*)carve((size_t)512 * 2048 * 4);
    unsigned short* h0bf   = (unsigned short*)carve((size_t)512 * 1024 * 2);
    float* outcat = (float*)carve((size_t)512 * 1024 * 4);
    unsigned short* Hbig   = (unsigned short*)carve((size_t)8192 * 4096 * 2); // struct then label
    float* Mmid   = (float*)carve((size_t)8192 * 1024 * 4);                   // struct then label
    (void)ws_size; (void)in_sizes; (void)n_in; (void)out_size;

    const float scale = 1.0f / 16384.0f;

    // 0) zero scalar accumulator
    zero_out_kernel<<<1, 1, 0, stream>>>(out);

    // 1) embedding -> bf16 x (512 x 256)
    embed_kernel<<<(512 * 256) / 256, 256, 0, stream>>>(word_inds, tag_inds, W_word, W_tag, x_bf);

    // 2) transpose all 4 Whh (2048 x 512) -> (512 x 2048)
    {
        int nb = (2048 * 512) / 256;
        transpose_kernel<<<nb, 256, 0, stream>>>(Whh_l0f, whhT0f, 2048, 512);
        transpose_kernel<<<nb, 256, 0, stream>>>(Whh_l0b, whhT0b, 2048, 512);
        transpose_kernel<<<nb, 256, 0, stream>>>(Whh_l1f, whhT1f, 2048, 512);
        transpose_kernel<<<nb, 256, 0, stream>>>(Whh_l1b, whhT1b, 2048, 512);
    }

    // 3) layer-0 input projections: xg = x @ Wih^T + b   (M=512, N=2048, K=256)
    {
        dim3 grid(2048 / 128, 512 / 128);
        convpad_kernel<<<(2048 * 256) / 256, 256, 0, stream>>>(Wih_l0f, wslot, 2048, 250, 256);
        gemm_bf16_kernel<<<grid, 256, 0, stream>>>(x_bf, wslot, b_l0f, xgA, 512, 2048, 256, 0);
        convpad_kernel<<<(2048 * 256) / 256, 256, 0, stream>>>(Wih_l0b, wslot, 2048, 250, 256);
        gemm_bf16_kernel<<<grid, 256, 0, stream>>>(x_bf, wslot, b_l0b, xgB, 512, 2048, 256, 0);
    }

    // 4) layer-0 recurrence (both directions concurrently) -> h0 bf16 (512 x 1024)
    lstm_rec2_kernel<<<2, 1024, 0, stream>>>(xgA, xgB, whhT0f, whhT0b, nullptr, h0bf, 1024, 512);

    // 5) layer-1 input projections: (M=512, N=2048, K=1024)
    {
        dim3 grid(2048 / 128, 512 / 128);
        convpad_kernel<<<(2048 * 1024) / 256, 256, 0, stream>>>(Wih_l1f, wslot, 2048, 1024, 1024);
        gemm_bf16_kernel<<<grid, 256, 0, stream>>>(h0bf, wslot, b_l1f, xgA, 512, 2048, 1024, 0);
        convpad_kernel<<<(2048 * 1024) / 256, 256, 0, stream>>>(Wih_l1b, wslot, 2048, 1024, 1024);
        gemm_bf16_kernel<<<grid, 256, 0, stream>>>(h0bf, wslot, b_l1b, xgB, 512, 2048, 1024, 0);
    }

    // 6) layer-1 recurrence -> outcat f32 (512 x 1024): fwd cols 0..511, back cols 512..1023
    lstm_rec2_kernel<<<2, 1024, 0, stream>>>(xgA, xgB, whhT1f, whhT1b, outcat, nullptr, 1024, 512);

    // 7) struct spans: gather -> Hs (8192 x 4096) bf16; GEMM+bias+relu -> Mmid; loss
    {
        int total = 8192 * 4096;
        span_gather_kernel<<<total / 256, 256, 0, stream>>>(outcat, s_lefts, s_rights, Hbig, 8192, 4);
        convpad_kernel<<<(1024 * 4096) / 256, 256, 0, stream>>>(Ws1, wslot, 1024, 4096, 4096);
        dim3 grid(1024 / 128, 8192 / 128);
        gemm_bf16_kernel<<<grid, 256, 0, stream>>>(Hbig, wslot, bs1, Mmid, 8192, 1024, 4096, 1);
        loss_kernel<<<8192, 256, 0, stream>>>(Mmid, Ws2, bs2, s_targets, out, 2, scale);
    }

    // 8) label spans: gather -> Hl (8192 x 3072) bf16; GEMM+bias+relu -> Mmid; loss
    {
        int total = 8192 * 3072;
        span_gather_kernel<<<total / 256, 256, 0, stream>>>(outcat, l_lefts, l_rights, Hbig, 8192, 3);
        convpad_kernel<<<(1024 * 3072) / 256, 256, 0, stream>>>(Wl1, wslot, 1024, 3072, 3072);
        dim3 grid(1024 / 128, 8192 / 128);
        gemm_bf16_kernel<<<grid, 256, 0, stream>>>(Hbig, wslot, bl1, Mmid, 8192, 1024, 3072, 1);
        loss_kernel<<<8192, 256, 0, stream>>>(Mmid, Wl2, bl2, l_targets, out, 56, scale);
    }
}
